// attention_net_3186865734129
// MI455X (gfx1250) — matrix operand order repaired and checked by reading, not compiled, not run on hardware
//
#include <hip/hip_runtime.h>

// ---------------------------------------------------------------------------
// Types for CDNA5 WMMA (wave32, 16x16x32 bf16 -> f32 accumulate)
// ---------------------------------------------------------------------------
typedef __bf16 bf16_t;
typedef bf16_t v16bf __attribute__((ext_vector_type(16)));
typedef float  v8f   __attribute__((ext_vector_type(8)));

union Frag { v16bf v; uint4 q[2]; };

__device__ __forceinline__ unsigned short f2bf(float f) {
    unsigned int u = __float_as_uint(f);
    unsigned int r = (u + 0x7FFFu + ((u >> 16) & 1u)) >> 16;  // round-to-nearest-even
    return (unsigned short)r;
}

// Async global->LDS 16-byte copy (CDNA5 GLOBAL_LOAD_ASYNC_TO_LDS_B128, GV mode).
// lds_off: low 32 bits of the generic shared address (== LDS byte offset).
__device__ __forceinline__ void async_copy16(unsigned lds_off, const void* gptr) {
    asm volatile("global_load_async_to_lds_b128 %0, %1, off"
                 :: "v"(lds_off), "v"(gptr)
                 : "memory");
}
// Wait until at most N async copies are still outstanding (in-order completion).
#define WAIT_ASYNC_LE(n) asm volatile("s_wait_asynccnt " #n ::: "memory")

__device__ __forceinline__ unsigned lds_off32(const void* p) {
    return (unsigned)(unsigned long long)p;
}

// ---------------------------------------------------------------------------
// Pack NCHW f32 activations into padded NHWC bf16 [B][16][16][C]
// Image pixel (y,x) lives at padded (y+1, x+1); border = 0 (pad=1 conv).
// ---------------------------------------------------------------------------
__global__ void pack_pad(const float* __restrict__ in, unsigned short* __restrict__ outp,
                         int C, int H, int W) {
    long idx = (long)blockIdx.x * blockDim.x + threadIdx.x;
    long total = (long)16 * 16 * 16 * C;
    if (idx >= total) return;
    int c = (int)(idx % C);
    long r = idx / C;
    int ix = (int)(r % 16); r /= 16;
    int iy = (int)(r % 16);
    int b  = (int)(r / 16);
    float v = 0.f;
    int y = iy - 1, x = ix - 1;
    if ((unsigned)y < (unsigned)H && (unsigned)x < (unsigned)W)
        v = in[(((long)b * C + c) * H + y) * W + x];
    outp[idx] = f2bf(v);
}

// ---------------------------------------------------------------------------
// Pack OIHW f32 weights (O=128, I=Cin, 3x3) into TRANSPOSED bf16 B-matrix
// BwT[oc][k], k = (ky*3+kx)*Cin + ic  (kpos-major, ic-minor)
// ---------------------------------------------------------------------------
__global__ void pack_w_T(const float* __restrict__ w, unsigned short* __restrict__ BwT, int Cin) {
    int K = 9 * Cin;
    long idx = (long)blockIdx.x * blockDim.x + threadIdx.x;
    long total = (long)128 * K;
    if (idx >= total) return;
    int oc = (int)(idx / K);
    int k  = (int)(idx % K);
    int kpos = k / Cin;
    int ic   = k - kpos * Cin;
    int ky = kpos / 3, kx = kpos % 3;
    BwT[idx] = f2bf(w[(((long)oc * Cin + ic) * 3 + ky) * 3 + kx]);
}

// ---------------------------------------------------------------------------
// Implicit-GEMM 3x3 conv via V_WMMA_F32_16X16X32_BF16 (all shapes compile-time).
//   M = 16*OH*OW, N = 128, K = 9*CIN (CIN is a power of two).
//   Block: 256 threads = 8 waves (2M x 4N); wave tile 32M x 32N (4 accums);
//   block tile 64M x 128N. LDS: double-buffered A[2][64][72], B^T[2][128][72].
//   Staging: GLOBAL_LOAD_ASYNC_TO_LDS_B128, double-buffered (issue tile i+1,
//   s_wait_asynccnt 6 for tile i — per-wave copies complete in order).
//   Fragments: ds_load_b128 pairs per the ISA 7.12.2 VGPR layouts.
// ---------------------------------------------------------------------------
template<int CIN, int OH, int OW, int STRIDE>
__global__ __launch_bounds__(256)
void conv_wmma(const unsigned short* __restrict__ Apad,   // [16][16][16][CIN] bf16
               const unsigned short* __restrict__ BwT,    // [128][9*CIN] bf16
               const float* __restrict__ bias,            // [128]
               float* __restrict__ out) {                 // [16][128][OH][OW] f32
    constexpr int S = OH * OW;
    constexpr int M = 16 * S;
    constexpr int K = 9 * CIN;
    constexpr int KI = K / 64;              // K iterations
    constexpr unsigned ABYTES = 64 * 72 * 2;
    constexpr unsigned BBYTES = 128 * 72 * 2;

    const int mblk = blockIdx.x;
    const int tid  = threadIdx.x;
    const int lane = tid & 31;
    const int wave = tid >> 5;
    const int wm   = wave & 1;             // 2 waves along M
    const int wn   = wave >> 1;            // 4 waves along N
    const int half = lane >> 4;
    const int lr   = lane & 15;

    __shared__ alignas(16) unsigned short A_lds[2][64 * 72];
    __shared__ alignas(16) unsigned short B_lds[2][128 * 72];

    v8f acc[2][2] = {};

    // --- A loader precompute: 512 x 16B units, 2 per thread ---
    int abase[2]; unsigned aldsoff[2];
    #pragma unroll
    for (int i = 0; i < 2; i++) {
        int u   = tid + i * 256;
        int r   = u >> 3;
        int seg = u & 7;
        int am  = mblk * 64 + r;
        if (am > M - 1) am = M - 1;        // clamp: junk rows are masked at store
        int ab  = am / S;  int rem = am - ab * S;
        int aoy = rem / OW; int aox = rem - aoy * OW;
        abase[i]   = ((ab * 16 + aoy * STRIDE) * 16 + aox * STRIDE) * CIN + seg * 8;
        aldsoff[i] = lds_off32(&A_lds[0][r * 72 + seg * 8]);
    }
    // --- B loader precompute: 1024 x 16B units, 4 per thread ---
    int bbase[4]; unsigned bldsoff[4];
    #pragma unroll
    for (int i = 0; i < 4; i++) {
        int u   = tid + i * 256;
        int n   = u >> 3;
        int seg = u & 7;
        bbase[i]   = n * K + seg * 8;
        bldsoff[i] = lds_off32(&B_lds[0][n * 72 + seg * 8]);
    }

    // stage a K-slab into LDS buffer `buf` (6 async copy instructions per wave)
    auto issue_tile = [&](int buf, int k0) {
        const int kpos = k0 / CIN;                 // CIN is 2^n -> shift
        const int ic0  = k0 - kpos * CIN;          // 64-chunk never crosses a tap
        const int ky   = kpos / 3, kx = kpos - ky * 3;
        const int aofs = (ky * 16 + kx) * CIN + ic0;
        #pragma unroll
        for (int i = 0; i < 2; i++)
            async_copy16(aldsoff[i] + buf * ABYTES, Apad + abase[i] + aofs);
        #pragma unroll
        for (int i = 0; i < 4; i++)
            async_copy16(bldsoff[i] + buf * BBYTES, BwT + bbase[i] + k0);
    };

    issue_tile(0, 0);
    for (int it = 0; it < KI; ++it) {
        const int buf = it & 1;
        if (it + 1 < KI) {
            __builtin_prefetch(BwT + (it + 2 < KI ? (it + 2) * 64 : 0), 0, 1);
            issue_tile(buf ^ 1, (it + 1) * 64);    // overlap DMA with WMMA
            WAIT_ASYNC_LE(6);                       // tile `it` landed; next 6 in flight
        } else {
            WAIT_ASYNC_LE(0);
        }
        __syncthreads();

        // ---- 2 k-substeps of 32; 4 WMMA each (2M x 2N register tile) ----
        #pragma unroll
        for (int ks = 0; ks < 64; ks += 32) {
            Frag af[2], bf[2];
            #pragma unroll
            for (int i = 0; i < 2; i++) {
                // A 16x32 bf16 layout: e<8 -> K=ks+8*half+e ; e>=8 -> +16
                const unsigned short* rp = &A_lds[buf][(wm * 32 + i * 16 + lr) * 72 + ks + half * 8];
                af[i].q[0] = *(const uint4*)rp;
                af[i].q[1] = *(const uint4*)(rp + 16);
            }
            #pragma unroll
            for (int j = 0; j < 2; j++) {
                // B 32x16 bf16 layout: e<8 -> K=ks+8*half+e ; e>=8 -> +16
                const unsigned short* rp = &B_lds[buf][(wn * 32 + j * 16 + lr) * 72 + ks + half * 8];
                bf[j].q[0] = *(const uint4*)rp;
                bf[j].q[1] = *(const uint4*)(rp + 16);
            }
            #pragma unroll
            for (int i = 0; i < 2; i++)
                #pragma unroll
                for (int j = 0; j < 2; j++)
                    acc[i][j] = __builtin_amdgcn_wmma_f32_16x16x32_bf16(
                        false, af[i].v, false, bf[j].v, (short)0, acc[i][j], false, false);
        }
        __syncthreads();   // everyone done reading buf before it is re-staged
    }

    // ---- store C (f32 16x16 layout: VGPR r -> M = r + 8*half, N = lane&15) ----
    #pragma unroll
    for (int i = 0; i < 2; i++) {
        int mbase = mblk * 64 + wm * 32 + i * 16 + half * 8;
        #pragma unroll
        for (int j = 0; j < 2; j++) {
            int n = wn * 32 + j * 16 + lr;
            float bv = bias[n];
            #pragma unroll
            for (int r = 0; r < 8; r++) {
                int m = mbase + r;
                if (m < M) {
                    int b = m / S; int rem = m - b * S;   // constant divisors
                    float v = fmaxf(acc[i][j][r] + bv, 0.f);
                    out[((long)(b * 128 + n)) * S + rem] = v;
                }
            }
        }
    }
}

// ---------------------------------------------------------------------------
// 1x1 "tidy" convs -> write rpn_score segment directly into d_out
// ---------------------------------------------------------------------------
__global__ void tidy1x1(const float* __restrict__ d, const float* __restrict__ w,
                        const float* __restrict__ bias, float* __restrict__ out,
                        int OC, int S, int boff, int bstride) {
    int idx = blockIdx.x * blockDim.x + threadIdx.x;
    int total = 16 * OC * S;
    if (idx >= total) return;
    int sp = idx % S; int r = idx / S;
    int oc = r % OC;  int b = r / OC;
    const float* dp = d + (long)b * 128 * S + sp;
    const float* wp = w + oc * 128;
    float acc = bias[oc];
    for (int c = 0; c < 128; c++) acc += dp[(long)c * S] * wp[c];
    out[(long)b * bstride + boff + oc * S + sp] = acc;
}

// ---------------------------------------------------------------------------
// Hard NMS top-6, one workgroup per batch element
// ---------------------------------------------------------------------------
__global__ __launch_bounds__(256)
void nms_top6(const float* __restrict__ scores, const int* __restrict__ anchors,
              int* __restrict__ idx_out, float* __restrict__ prob_out) {
    const int NA = 1614;
    int b = blockIdx.x;
    int tid = threadIdx.x;
    __shared__ float sc[NA];
    __shared__ float rv[256];
    __shared__ int   ri[256];
    __shared__ int   bestIdx;

    for (int i = tid; i < NA; i += 256) sc[i] = scores[(long)b * NA + i];
    __syncthreads();

    for (int t = 0; t < 6; t++) {
        float mv = -3.402823466e38f; int mi = NA;
        for (int i = tid; i < NA; i += 256) {
            float v = sc[i];
            if (v > mv) { mv = v; mi = i; }   // strictly-greater keeps first index
        }
        rv[tid] = mv; ri[tid] = mi;
        __syncthreads();
        for (int s = 128; s > 0; s >>= 1) {
            if (tid < s) {
                float v2 = rv[tid + s]; int i2 = ri[tid + s];
                if (v2 > rv[tid] || (v2 == rv[tid] && i2 < ri[tid])) { rv[tid] = v2; ri[tid] = i2; }
            }
            __syncthreads();
        }
        if (tid == 0) {
            bestIdx = ri[0];
            idx_out[b * 6 + t]  = ri[0];
            prob_out[b * 6 + t] = rv[0];
        }
        __syncthreads();
        int j = bestIdx;
        float y0 = (float)anchors[j * 4 + 0], x0 = (float)anchors[j * 4 + 1];
        float y1 = (float)anchors[j * 4 + 2], x1 = (float)anchors[j * 4 + 3];
        float a1 = (y1 - y0) * (x1 - x0);
        for (int i = tid; i < NA; i += 256) {
            float by0 = (float)anchors[i * 4 + 0], bx0 = (float)anchors[i * 4 + 1];
            float by1 = (float)anchors[i * 4 + 2], bx1 = (float)anchors[i * 4 + 3];
            float yy0 = fmaxf(y0, by0), xx0 = fmaxf(x0, bx0);
            float yy1 = fminf(y1, by1), xx1 = fminf(x1, bx1);
            float inter = fmaxf(yy1 - yy0, 0.f) * fmaxf(xx1 - xx0, 0.f);
            float a2 = (by1 - by0) * (bx1 - bx0);
            float iou = inter / (a1 + a2 - inter);
            if (iou > 0.25f) sc[i] = -3.402823466e38f;
        }
        __syncthreads();
    }
}

// ---------------------------------------------------------------------------
// Crop + bilinear resize 224x224 from virtually padded (PAD=224) image
// ---------------------------------------------------------------------------
__device__ __forceinline__ float sample_pad(const float* __restrict__ x, int b, int c,
                                            int yi, int xi) {
    yi -= 224; xi -= 224;
    if ((unsigned)yi < 448u && (unsigned)xi < 448u)
        return x[(((long)b * 3 + c) * 448 + yi) * 448 + xi];
    return 0.f;
}

__global__ void crop_resize(const float* __restrict__ x, const int* __restrict__ anchors,
                            const int* __restrict__ sel, float* __restrict__ out) {
    long idx = (long)blockIdx.x * blockDim.x + threadIdx.x;
    const long total = (long)16 * 6 * 3 * 224 * 224;
    if (idx >= total) return;
    int px = (int)(idx % 224); long r = idx / 224;
    int py = (int)(r % 224); r /= 224;
    int c  = (int)(r % 3);   r /= 3;
    int t  = (int)(r % 6);
    int b  = (int)(r / 6);

    int j = sel[b * 6 + t];
    float y0 = (float)anchors[j * 4 + 0], x0 = (float)anchors[j * 4 + 1];
    float y1 = (float)anchors[j * 4 + 2], x1 = (float)anchors[j * 4 + 3];

    float ty = (float)py * (1.f / 223.f);
    float tx = (float)px * (1.f / 223.f);
    float ys = y0 + ty * (y1 - y0 - 1.f);
    float xs = x0 + tx * (x1 - x0 - 1.f);
    int y0i = (int)floorf(ys);
    int x0i = (int)floorf(xs);
    int y1i = min(y0i + 1, 895);
    int x1i = min(x0i + 1, 895);
    float wy = ys - (float)y0i, wx = xs - (float)x0i;

    float tl = sample_pad(x, b, c, y0i, x0i);
    float tr = sample_pad(x, b, c, y0i, x1i);
    float bl = sample_pad(x, b, c, y1i, x0i);
    float br = sample_pad(x, b, c, y1i, x1i);
    out[idx] = tl * (1.f - wy) * (1.f - wx) + tr * (1.f - wy) * wx
             + bl * wy * (1.f - wx) + br * wy * wx;
}

// ---------------------------------------------------------------------------
// Host-side launcher
// ---------------------------------------------------------------------------
extern "C" void kernel_launch(void* const* d_in, const int* in_sizes, int n_in,
                              void* d_out, int out_size, void* d_ws, size_t ws_size,
                              hipStream_t stream) {
    const float* x        = (const float*)d_in[0];   // (16,3,448,448)
    const float* rpn_feat = (const float*)d_in[1];   // (16,2048,14,14)
    const int*   anchors  = (const int*)  d_in[2];   // (1614,4)
    const float* w1 = (const float*)d_in[3];  const float* b1 = (const float*)d_in[4];
    const float* w2 = (const float*)d_in[5];  const float* b2 = (const float*)d_in[6];
    const float* w3 = (const float*)d_in[7];  const float* b3 = (const float*)d_in[8];
    const float* tw1 = (const float*)d_in[9];  const float* tb1 = (const float*)d_in[10];
    const float* tw2 = (const float*)d_in[11]; const float* tb2 = (const float*)d_in[12];
    const float* tw3 = (const float*)d_in[13]; const float* tb3 = (const float*)d_in[14];

    float* out = (float*)d_out;
    // d_out layout: rpn_score(25824) | top_n_index(96, int bits) | top_n_prob(96) | part_imgs
    int*   idx_out  = (int*)(out + 25824);
    float* prob_out = out + 25920;
    float* imgs_out = out + 26016;

    // workspace carve-out (256 B aligned); A2/A3 and Bw2/Bw3 reuse conv1 regions
    unsigned char* p = (unsigned char*)d_ws;
    auto alloc = [&](size_t bytes) { void* r = (void*)p; p += (bytes + 255) & ~(size_t)255; return r; };
    unsigned short* A1  = (unsigned short*)alloc((size_t)16 * 16 * 16 * 2048 * 2); // 16.78 MB
    unsigned short* Bw1 = (unsigned short*)alloc((size_t)128 * 18432 * 2);         //  4.72 MB
    float* d1 = (float*)alloc((size_t)16 * 128 * 196 * 4);
    float* d2 = (float*)alloc((size_t)16 * 128 * 49 * 4);
    float* d3 = (float*)alloc((size_t)16 * 128 * 16 * 4);
    unsigned short* A23  = A1;    // reused after conv1 completes (stream-ordered)
    unsigned short* Bw23 = Bw1;

    const int T = 256;
    auto blocks = [](long total, int th) { return (unsigned)((total + th - 1) / th); };

    // stage 1: conv 2048->128 @14x14 (the compute-heavy GEMM: M=3136, K=18432)
    pack_pad<<<blocks((long)16 * 16 * 16 * 2048, T), T, 0, stream>>>(rpn_feat, A1, 2048, 14, 14);
    pack_w_T<<<blocks((long)128 * 9 * 2048, T),      T, 0, stream>>>(w1, Bw1, 2048);
    conv_wmma<2048, 14, 14, 1><<<49, T, 0, stream>>>(A1, Bw1, b1, d1);   // 3136/64 = 49

    // stage 2: conv 128->128 s2 @7x7 (M=784)
    pack_pad<<<blocks((long)16 * 16 * 16 * 128, T), T, 0, stream>>>(d1, A23, 128, 14, 14);
    pack_w_T<<<blocks((long)128 * 9 * 128, T),      T, 0, stream>>>(w2, Bw23, 128);
    conv_wmma<128, 7, 7, 2><<<13, T, 0, stream>>>(A23, Bw23, b2, d2);    // ceil(784/64)

    // stage 3: conv 128->128 s2 @4x4 (M=256)
    pack_pad<<<blocks((long)16 * 16 * 16 * 128, T), T, 0, stream>>>(d2, A23, 128, 7, 7);
    pack_w_T<<<blocks((long)128 * 9 * 128, T),      T, 0, stream>>>(w3, Bw23, 128);
    conv_wmma<128, 4, 4, 2><<<4, T, 0, stream>>>(A23, Bw23, b3, d3);     // 256/64

    // tidy 1x1 convs -> rpn_score segments (per-batch stride 1614)
    tidy1x1<<<blocks((long)16 * 6 * 196, T), T, 0, stream>>>(d1, tw1, tb1, out, 6, 196, 0,    1614);
    tidy1x1<<<blocks((long)16 * 6 * 49,  T), T, 0, stream>>>(d2, tw2, tb2, out, 6, 49,  1176, 1614);
    tidy1x1<<<blocks((long)16 * 9 * 16,  T), T, 0, stream>>>(d3, tw3, tb3, out, 9, 16,  1470, 1614);

    // hard NMS top-6 per batch
    nms_top6<<<16, T, 0, stream>>>(out, anchors, idx_out, prob_out);

    // crop + bilinear resize (virtual PAD=224, no materialized padded image)
    crop_resize<<<blocks((long)16 * 6 * 3 * 224 * 224, T), T, 0, stream>>>(x, anchors, idx_out, imgs_out);
}
